// Receiver_10462540333589
// MI455X (gfx1250) — matrix operand order repaired and checked by reading, not compiled, not run on hardware
//
#include <hip/hip_runtime.h>
#include <hip/hip_bf16.h>
#include <stdint.h>
#include <stddef.h>

// Problem dims (match reference)
#define B_SZ   512
#define T_SZ   64
#define EMBD   512
#define HID    1024
#define NIMG   2048
#define G4H    4096    // 4*HID
#define KCAT   1536    // EMBD + HID

// WMMA types (CDNA5 wave32)
typedef __bf16 v16bf __attribute__((ext_vector_type(16)));
typedef float  v8f   __attribute__((ext_vector_type(8)));
typedef unsigned int v4u __attribute__((ext_vector_type(4)));

union Frag16 { v4u q[2]; v16bf v; };

__device__ __forceinline__ unsigned short f32_to_bf16_rn(float f) {
  unsigned int u = __float_as_uint(f);
  if ((u & 0x7fffffffu) > 0x7f800000u) return (unsigned short)((u >> 16) | 0x0040u); // quiet NaN
  u += 0x7fffu + ((u >> 16) & 1u); // round to nearest even
  return (unsigned short)(u >> 16);
}

__device__ __forceinline__ float sigmoidf_fast(float x) {
  return 1.0f / (1.0f + __expf(-x));
}

// --- CDNA5 async global->LDS copy (GLOBAL_LOAD_ASYNC_TO_LDS_B128, ASYNCcnt) ---
__device__ __forceinline__ void async_load_b128(const void* gptr, unsigned lds_byte_off) {
  // VDST = per-lane LDS byte offset, VADDR = 64-bit global address (GV mode)
  asm volatile("global_load_async_to_lds_b128 %0, %1, off"
               :: "v"(lds_byte_off), "v"(gptr) : "memory");
}

__device__ __forceinline__ void wait_async_zero() {
#if __has_builtin(__builtin_amdgcn_s_wait_asynccnt)
  __builtin_amdgcn_s_wait_asynccnt(0);
#else
  asm volatile("s_wait_asynccnt 0x0" ::: "memory");
#endif
}

// ---------------------------------------------------------------------------
// bf16 WMMA GEMM:  D[M,N] = A[M,K] * Bw[N,K]^T + bias[N]
// A, Bw row-major bf16 (K contiguous) -> both fragments load contiguous K.
// Block: 256 threads = 8 waves; tile BM=128 x BN=128, BK=32 (one WMMA K-step).
// Wave grid 4(M) x 2(N); each wave computes 32x64 = 2x4 v_wmma accumulators.
// Double-buffered LDS filled by async-to-LDS copies; one barrier per K-step.
// ---------------------------------------------------------------------------
#define BM 128
#define BN 128
#define BK 32
#define LDSS 40   // padded row stride in ushorts (80B: 16B-aligned, conflict-free)
#define TILE_USH (BM * LDSS)                   // 5120 ushorts per tile
#define GEMM_LDS_BYTES (4 * TILE_USH * 2)      // A0,A1,B0,B1 = 40960 B

__global__ __launch_bounds__(256)
void gemm_bf16_wmma(const unsigned short* __restrict__ A, int lda,
                    const unsigned short* __restrict__ Bw, int ldb,
                    float* __restrict__ D, int ldd,
                    const float* __restrict__ bias, int K) {
  extern __shared__ unsigned short smem[];  // dynamic LDS => byte offsets from 0

  const int tid   = threadIdx.x;
  const int lane  = tid & 31;
  const int wid   = tid >> 5;     // 0..7
  const int waveM = wid & 3;      // 4 M-slots of 32 rows
  const int waveN = wid >> 2;     // 2 N-slots of 64 cols
  const int mBase = blockIdx.y * BM;
  const int nBase = blockIdx.x * BN;
  const int half  = lane >> 4;    // 0/1
  const int l16   = lane & 15;

  // Accumulators: each fragment's 8 elements share the same output column,
  // so bias can be splatted into the accumulator init.
  v8f acc[2][4];
  #pragma unroll
  for (int ni = 0; ni < 4; ++ni) {
    float bv = bias[nBase + waveN * 64 + ni * 16 + l16];
    #pragma unroll
    for (int mi = 0; mi < 2; ++mi)
      #pragma unroll
      for (int e = 0; e < 8; ++e) acc[mi][ni][e] = bv;
  }

  // Cooperative tile fill: 512 (row, 16B-seg) units over 2 iterations/thread.
  const int ldRow = tid >> 2;        // 0..63
  const int ldSeg = (tid & 3) * 8;   // ushort offset within row (x8 = 16B)

  // LDS tile base offsets (ushort units): A0, A1, B0, B1
  const unsigned offA[2] = { 0u, (unsigned)TILE_USH };
  const unsigned offB[2] = { 2u * TILE_USH, 3u * TILE_USH };

  auto issue_tile = [&](int k0, int buf) {
    #pragma unroll
    for (int it = 0; it < 2; ++it) {
      const int row = ldRow + it * 64;
      const unsigned lofs = (unsigned)(row * LDSS + ldSeg);
      async_load_b128(A  + (size_t)(mBase + row) * lda + k0 + ldSeg,
                      (offA[buf] + lofs) * 2u);
      async_load_b128(Bw + (size_t)(nBase + row) * ldb + k0 + ldSeg,
                      (offB[buf] + lofs) * 2u);
    }
  };

  const int nk = K / BK;
  issue_tile(0, 0);
  wait_async_zero();
  __syncthreads();

  for (int kt = 0; kt < nk; ++kt) {
    const int buf = kt & 1;
    if (kt + 1 < nk) issue_tile((kt + 1) * BK, buf ^ 1);  // overlap with compute
    if (kt + 2 < nk) {  // warm L2 two tiles ahead (global_prefetch_b8)
      __builtin_prefetch(A  + (size_t)(mBase + ldRow) * lda + (kt + 2) * BK + ldSeg, 0, 1);
      __builtin_prefetch(Bw + (size_t)(nBase + ldRow) * ldb + (kt + 2) * BK + ldSeg, 0, 1);
    }

    const unsigned short* Ab = &smem[offA[buf]];
    const unsigned short* Bb = &smem[offB[buf]];

    // B fragments (32x16, K x N): lane = col n (l16); halves hold K 0..15 / 16..31.
    Frag16 bf[4];
    #pragma unroll
    for (int ni = 0; ni < 4; ++ni) {
      const int n = waveN * 64 + ni * 16 + l16;
      const unsigned short* p = &Bb[n * LDSS + half * 16];
      bf[ni].q[0] = *(const v4u*)(p);
      bf[ni].q[1] = *(const v4u*)(p + 8);
    }
    // A fragments (16x32, M x K): lane = row m (l16);
    // elems 0..15 -> K = half*16+0..15.
    #pragma unroll
    for (int mi = 0; mi < 2; ++mi) {
      const int m = waveM * 32 + mi * 16 + l16;
      Frag16 af;
      const unsigned short* p = &Ab[m * LDSS + half * 16];
      af.q[0] = *(const v4u*)(p);
      af.q[1] = *(const v4u*)(p + 8);
      #pragma unroll
      for (int ni = 0; ni < 4; ++ni) {
        acc[mi][ni] = __builtin_amdgcn_wmma_f32_16x16x32_bf16(
            false, af.v, false, bf[ni].v, (short)0, acc[mi][ni], false, false);
      }
    }

    wait_async_zero();   // my async fills of buf^1 have landed
    __syncthreads();     // everyone done reading buf / filling buf^1
  }

  // C/D layout: VGPR r -> row (8*half + r), col = l16.
  #pragma unroll
  for (int mi = 0; mi < 2; ++mi) {
    const int mrow = mBase + waveM * 32 + mi * 16 + half * 8;
    #pragma unroll
    for (int ni = 0; ni < 4; ++ni) {
      const int col = nBase + waveN * 64 + ni * 16 + l16;
      float* dp = D + (size_t)mrow * ldd + col;
      #pragma unroll
      for (int r = 0; r < 8; ++r) dp[(size_t)r * ldd] = acc[mi][ni][r];
    }
  }
}

// ---------------------------------------------------------------------------
// Prep kernels
// ---------------------------------------------------------------------------
__global__ void build_wcat(const float* __restrict__ Wih,
                           const float* __restrict__ Whh,
                           unsigned short* __restrict__ Wcat) {
  int idx = blockIdx.x * blockDim.x + threadIdx.x;   // < G4H*KCAT
  int n = idx / KCAT, k = idx % KCAT;
  float w = (k < EMBD) ? Wih[(size_t)n * EMBD + k]
                       : Whh[(size_t)n * HID + (k - EMBD)];
  Wcat[idx] = f32_to_bf16_rn(w);
}

__global__ void cvt_bf16(const float* __restrict__ src,
                         unsigned short* __restrict__ dst, int n) {
  int idx = blockIdx.x * blockDim.x + threadIdx.x;
  if (idx < n) dst[idx] = f32_to_bf16_rn(src[idx]);
}

__global__ void bias_sum(const float* __restrict__ bih,
                         const float* __restrict__ bhh,
                         float* __restrict__ bsum) {
  int idx = blockIdx.x * blockDim.x + threadIdx.x;   // < G4H
  bsum[idx] = bih[idx] + bhh[idx];
}

// X[t*B + b, e] = bf16(emb[m[b,t], e])   (time-major so each step is contiguous)
__global__ void embed_gather(const int* __restrict__ m,
                             const float* __restrict__ emb,
                             unsigned short* __restrict__ X) {
  int idx = blockIdx.x * blockDim.x + threadIdx.x;   // < T*B*E
  int e   = idx & (EMBD - 1);
  int tb  = idx / EMBD;
  int b   = tb & (B_SZ - 1);
  int t   = tb / B_SZ;
  int tok = m[b * T_SZ + t];
  X[idx]  = f32_to_bf16_rn(emb[(size_t)tok * EMBD + e]);
}

// c = 0; h-part of Acat = 0 (h0 == 0)
__global__ void init_state(float* __restrict__ c, unsigned short* __restrict__ Acat) {
  int idx = blockIdx.x * blockDim.x + threadIdx.x;   // < B*H
  int b = idx >> 10, j = idx & (HID - 1);
  c[idx] = 0.0f;
  Acat[(size_t)b * KCAT + EMBD + j] = 0;
}

// Copy x_t (already bf16, contiguous [B,E]) into Acat columns [0,E)
__global__ void pack_x(const unsigned short* __restrict__ Xt,
                       unsigned short* __restrict__ Acat) {
  int idx = blockIdx.x * blockDim.x + threadIdx.x;   // < B*E/8
  int row = idx >> 6;            // /(E/8)
  int seg = (idx & 63) * 8;
  *(v4u*)(&Acat[(size_t)row * KCAT + seg]) = *(const v4u*)(&Xt[(size_t)row * EMBD + seg]);
}

// LSTM pointwise: gates [B,4H] fp32 (i,f,g,o order), update c, emit h (bf16)
// into Acat's h columns (feeds next step) and into hbf (feeds output GEMM).
__global__ void lstm_pointwise(const float* __restrict__ gates,
                               float* __restrict__ c,
                               unsigned short* __restrict__ Acat,
                               unsigned short* __restrict__ hbf) {
  int idx = blockIdx.x * blockDim.x + threadIdx.x;   // < B*H
  int b = idx >> 10, j = idx & (HID - 1);
  const float* g = gates + (size_t)b * G4H;
  float gi = sigmoidf_fast(g[j]);
  float gf = sigmoidf_fast(g[HID + j]);
  float gg = tanhf(g[2 * HID + j]);
  float go = sigmoidf_fast(g[3 * HID + j]);
  float cn = gf * c[idx] + gi * gg;
  c[idx] = cn;
  unsigned short hb = f32_to_bf16_rn(go * tanhf(cn));
  Acat[(size_t)b * KCAT + EMBD + j] = hb;
  hbf[idx] = hb;
}

// ---------------------------------------------------------------------------
// Host side
// ---------------------------------------------------------------------------
static inline size_t alignUp(size_t x) { return (x + 255) & ~(size_t)255; }

extern "C" void kernel_launch(void* const* d_in, const int* in_sizes, int n_in,
                              void* d_out, int out_size, void* d_ws, size_t ws_size,
                              hipStream_t stream) {
  (void)in_sizes; (void)n_in; (void)out_size; (void)ws_size;
  const int*   m     = (const int*)  d_in[0];
  const float* emb   = (const float*)d_in[1];
  const float* W_ih  = (const float*)d_in[2];
  const float* W_hh  = (const float*)d_in[3];
  const float* b_ih  = (const float*)d_in[4];
  const float* b_hh  = (const float*)d_in[5];
  const float* W_out = (const float*)d_in[6];
  const float* b_out = (const float*)d_in[7];
  float* out = (float*)d_out;

  // Workspace carve-up (~64 MB total)
  char* ws = (char*)d_ws;
  size_t off = 0;
  unsigned short* Wcat  = (unsigned short*)(ws + off); off = alignUp(off + (size_t)G4H * KCAT * 2);
  unsigned short* Woutb = (unsigned short*)(ws + off); off = alignUp(off + (size_t)NIMG * HID * 2);
  float*          bsum  = (float*)         (ws + off); off = alignUp(off + (size_t)G4H * 4);
  unsigned short* Xb    = (unsigned short*)(ws + off); off = alignUp(off + (size_t)T_SZ * B_SZ * EMBD * 2);
  unsigned short* Acat  = (unsigned short*)(ws + off); off = alignUp(off + (size_t)B_SZ * KCAT * 2);
  float*          gates = (float*)         (ws + off); off = alignUp(off + (size_t)B_SZ * G4H * 4);
  float*          cst   = (float*)         (ws + off); off = alignUp(off + (size_t)B_SZ * HID * 4);
  unsigned short* hbf   = (unsigned short*)(ws + off); off = alignUp(off + (size_t)B_SZ * HID * 2);

  // Prep
  build_wcat  <<<(G4H * KCAT) / 256, 256, 0, stream>>>(W_ih, W_hh, Wcat);
  cvt_bf16    <<<(NIMG * HID) / 256, 256, 0, stream>>>(W_out, Woutb, NIMG * HID);
  bias_sum    <<<G4H / 256, 256, 0, stream>>>(b_ih, b_hh, bsum);
  embed_gather<<<(T_SZ * B_SZ * EMBD) / 256, 256, 0, stream>>>(m, emb, Xb);
  init_state  <<<(B_SZ * HID) / 256, 256, 0, stream>>>(cst, Acat);

  // Recurrence: 64 sequential steps, each one GEMM + pointwise on `stream`
  const dim3 gemmGridStep(G4H / BN, B_SZ / BM);   // 32 x 4
  for (int t = 0; t < T_SZ; ++t) {
    pack_x<<<(B_SZ * EMBD / 8) / 256, 256, 0, stream>>>(Xb + (size_t)t * B_SZ * EMBD, Acat);
    gemm_bf16_wmma<<<gemmGridStep, 256, GEMM_LDS_BYTES, stream>>>(
        Acat, KCAT, Wcat, KCAT, gates, G4H, bsum, KCAT);
    lstm_pointwise<<<(B_SZ * HID) / 256, 256, 0, stream>>>(gates, cst, Acat, hbf);
  }

  // Output projection: out = h @ W_out^T + b_out
  const dim3 gemmGridOut(NIMG / BN, B_SZ / BM);   // 16 x 4
  gemm_bf16_wmma<<<gemmGridOut, 256, GEMM_LDS_BYTES, stream>>>(
      hbf, HID, Woutb, HID, out, NIMG, b_out, HID);
}
